// EfficientViTBlock_44023414784419
// MI455X (gfx1250) — compile-verified, matrix operand order checked by reading, not run on hardware
//
#include <hip/hip_runtime.h>

// ---------------------------------------------------------------------------
// CDNA5 (gfx1250) EfficientViT block. All 1x1 convs + attention GEMMs run on
// v_wmma_f32_16x16x32_bf16 (wave32, fp32 accumulate). Depthwise convs on VALU.
// Main GEMM: block tile 32(M)x128(N), each wave owns 16x32 via two
// accumulators sharing one A fragment (2 WMMAs per LDS fragment fetch).
// ---------------------------------------------------------------------------

typedef __attribute__((ext_vector_type(16))) __bf16        v16bf;
typedef __attribute__((ext_vector_type(8)))  float         v8f;
typedef __attribute__((ext_vector_type(8)))  unsigned int  v8u;

#define BB   16
#define HW   1024

__device__ __forceinline__ unsigned short f2bf(float f) {
    unsigned int u = __builtin_bit_cast(unsigned int, f);
    u += 0x7FFFu + ((u >> 16) & 1u);          // round-to-nearest-even
    return (unsigned short)(u >> 16);
}
__device__ __forceinline__ float bf2f(unsigned short h) {
    unsigned int u = ((unsigned int)h) << 16;
    return __builtin_bit_cast(float, u);
}
// K index of the even half of pair p for this lane-half (CDNA5 16-bit A/B layout):
// lanes 0-15: halves 0-7 -> K0-7, halves 8-15 -> K16-23 ; lanes 16-31: +8.
__device__ __forceinline__ int kpair(int p, int hi) {
    return ((p < 4) ? (2 * p) : (16 + 2 * (p - 4))) + hi * 8;
}

// ------------------------------- fp32 -> bf16 ------------------------------
__global__ void f2bf_kernel(const float* __restrict__ in,
                            unsigned short* __restrict__ out, int n) {
    int i = blockIdx.x * 256 + threadIdx.x;
    if (i < n) out[i] = f2bf(in[i]);
}

// --------------------------- generic bf16 WMMA GEMM ------------------------
// out[b][m][n] = sum_k W[m][k] * X[b][k][n],  N fixed = 1024 (H*W)
// MODE 0: store bf16                      (qkv)
// MODE 1: BN + residual -> f32 + bf16     (proj)
// MODE 2: +bias, hardswish -> bf16        (mb1)
// MODE 3: BN + residual -> f32 (d_out)    (pw)
template <int MODE>
__global__ __launch_bounds__(256)
void gemm_bf16_kernel(const unsigned short* __restrict__ W,
                      const unsigned short* __restrict__ X,
                      int M, int K,
                      unsigned short* __restrict__ out_bf,
                      float* __restrict__ out_f,
                      const float* __restrict__ resid,
                      const float* __restrict__ p0, const float* __restrict__ p1,
                      const float* __restrict__ p2, const float* __restrict__ p3) {
    __shared__ unsigned short As[32][34];    // [m][k], pad keeps u32 pair reads aligned
    __shared__ unsigned short Bs[128][34];   // [n][k] (transposed so K-pairs contiguous)

    const int tid  = threadIdx.x;
    const int wid  = tid >> 5, lane = tid & 31;
    const int hi   = lane >> 4, r   = lane & 15;
    const int wr   = wid >> 2,  wc  = wid & 3;   // wave tile: 16M x 32N
    const int b    = blockIdx.z;
    const int mbase = blockIdx.y * 32;
    const int nbase = blockIdx.x * 128;
    const unsigned short* Xb = X + (size_t)b * K * HW;

    v8f acc0 = {}, acc1 = {};
    for (int kb = 0; kb < K; kb += 32) {
        __syncthreads();
        {   // A tile 32x32, 4 elems/thread
            int e = tid * 4, m = e >> 5, k = e & 31;
            const unsigned short* src = W + (size_t)(mbase + m) * K + kb + k;
            As[m][k]     = src[0]; As[m][k + 1] = src[1];
            As[m][k + 2] = src[2]; As[m][k + 3] = src[3];
        }
        {   // B tile 32x128 -> transposed LDS, 16 elems/thread (coalesced in n)
            int e = tid * 16, k = e >> 7, n = e & 127;
            const unsigned short* src = Xb + (size_t)(kb + k) * HW + nbase + n;
#pragma unroll
            for (int j = 0; j < 16; ++j) Bs[n + j][k] = src[j];
        }
        __syncthreads();
        v8u au, b0u, b1u;
#pragma unroll
        for (int p = 0; p < 8; ++p) {
            int k = kpair(p, hi);
            au[p]  = *(const unsigned int*)&As[wr * 16 + r][k];
            b0u[p] = *(const unsigned int*)&Bs[wc * 32 + r][k];
            b1u[p] = *(const unsigned int*)&Bs[wc * 32 + 16 + r][k];
        }
        v16bf a = __builtin_bit_cast(v16bf, au);
        acc0 = __builtin_amdgcn_wmma_f32_16x16x32_bf16(false, a, false,
                  __builtin_bit_cast(v16bf, b0u), (short)0, acc0, false, false);
        acc1 = __builtin_amdgcn_wmma_f32_16x16x32_bf16(false, a, false,
                  __builtin_bit_cast(v16bf, b1u), (short)0, acc1, false, false);
    }
    const int n0 = nbase + wc * 32 + r;
#pragma unroll
    for (int i = 0; i < 8; ++i) {
        int m = mbase + wr * 16 + i + 8 * hi;
        size_t o0 = ((size_t)b * M + m) * HW + n0;
        float v0 = acc0[i], v1 = acc1[i];
        if (MODE == 0) {
            out_bf[o0]      = f2bf(v0);
            out_bf[o0 + 16] = f2bf(v1);
        } else if (MODE == 1 || MODE == 3) {
            float s = p0[m] * rsqrtf(p3[m] + 1e-5f);
            float t = p1[m] - p2[m] * s;
            v0 = v0 * s + t + resid[o0];
            v1 = v1 * s + t + resid[o0 + 16];
            if (MODE == 1) {
                out_f[o0] = v0;      out_f[o0 + 16] = v1;
                out_bf[o0] = f2bf(v0); out_bf[o0 + 16] = f2bf(v1);
            } else {
                out_f[o0] = v0;      out_f[o0 + 16] = v1;
            }
        } else { // MODE 2: bias + hardswish
            v0 += p0[m]; v1 += p0[m];
            float c0 = fminf(fmaxf(v0 + 3.0f, 0.0f), 6.0f);
            float c1 = fminf(fmaxf(v1 + 3.0f, 0.0f), 6.0f);
            out_bf[o0]      = f2bf(v0 * c0 * (1.0f / 6.0f));
            out_bf[o0 + 16] = f2bf(v1 * c1 * (1.0f / 6.0f));
        }
    }
}

// ---------------------- grouped 32x32 pointwise conv (WMMA) ----------------
__global__ __launch_bounds__(256)
void gemm_group32_kernel(const unsigned short* __restrict__ Wg,  // [768][32]
                         const unsigned short* __restrict__ X,   // [B][768][1024]
                         unsigned short* __restrict__ out) {     // [B][768][1024]
    __shared__ unsigned short As[32][34];
    __shared__ unsigned short Bs[128][34];
    const int tid = threadIdx.x, wid = tid >> 5, lane = tid & 31;
    const int hi = lane >> 4, r = lane & 15;
    const int wr = wid >> 2, wc = wid & 3;
    const int b = blockIdx.z, g = blockIdx.y, nbase = blockIdx.x * 128;
    {
        int e = tid * 4, m = e >> 5, k = e & 31;
        const unsigned short* src = Wg + (size_t)(g * 32 + m) * 32 + k;
        As[m][k] = src[0]; As[m][k + 1] = src[1];
        As[m][k + 2] = src[2]; As[m][k + 3] = src[3];
    }
    {
        int e = tid * 16, k = e >> 7, n = e & 127;
        const unsigned short* src = X + ((size_t)b * 768 + g * 32 + k) * HW + nbase + n;
#pragma unroll
        for (int j = 0; j < 16; ++j) Bs[n + j][k] = src[j];
    }
    __syncthreads();
    v8u au, b0u, b1u;
#pragma unroll
    for (int p = 0; p < 8; ++p) {
        int k = kpair(p, hi);
        au[p]  = *(const unsigned int*)&As[wr * 16 + r][k];
        b0u[p] = *(const unsigned int*)&Bs[wc * 32 + r][k];
        b1u[p] = *(const unsigned int*)&Bs[wc * 32 + 16 + r][k];
    }
    v16bf a = __builtin_bit_cast(v16bf, au);
    v8f acc0 = {}, acc1 = {};
    acc0 = __builtin_amdgcn_wmma_f32_16x16x32_bf16(false, a, false,
              __builtin_bit_cast(v16bf, b0u), (short)0, acc0, false, false);
    acc1 = __builtin_amdgcn_wmma_f32_16x16x32_bf16(false, a, false,
              __builtin_bit_cast(v16bf, b1u), (short)0, acc1, false, false);
#pragma unroll
    for (int i = 0; i < 8; ++i) {
        int co = g * 32 + wr * 16 + i + 8 * hi;
        size_t o = ((size_t)b * 768 + co) * HW + nbase + wc * 32 + r;
        out[o]      = f2bf(acc0[i]);
        out[o + 16] = f2bf(acc1[i]);
    }
}

// ------------------------------ depthwise 5x5 ------------------------------
__global__ __launch_bounds__(256)
void dw5x5_kernel(const unsigned short* __restrict__ in, const float* __restrict__ w,
                  unsigned short* __restrict__ out) {
    int bc  = blockIdx.x;                 // b*768 + c
    int c   = bc % 768;
    int pix = blockIdx.y * 256 + threadIdx.x;
    int y = pix >> 5, x = pix & 31;
    const unsigned short* ip = in + (size_t)bc * HW;
    const float* wp = w + c * 25;
    float acc = 0.f;
#pragma unroll
    for (int dy = 0; dy < 5; ++dy) {
        int yy = y + dy - 2;
        if (yy < 0 || yy > 31) continue;
#pragma unroll
        for (int dx = 0; dx < 5; ++dx) {
            int xx = x + dx - 2;
            if (xx < 0 || xx > 31) continue;
            acc += bf2f(ip[yy * 32 + xx]) * wp[dy * 5 + dx];
        }
    }
    out[(size_t)bc * HW + pix] = f2bf(acc);
}

// ------------------------ depthwise 3x3 + bias + hswish --------------------
__global__ __launch_bounds__(256)
void dw3x3_kernel(const unsigned short* __restrict__ in, const float* __restrict__ w,
                  const float* __restrict__ bias, unsigned short* __restrict__ out) {
    int bc  = blockIdx.x;                 // b*1024 + c
    int c   = bc & 1023;
    int pix = blockIdx.y * 256 + threadIdx.x;
    int y = pix >> 5, x = pix & 31;
    const unsigned short* ip = in + (size_t)bc * HW;
    const float* wp = w + c * 9;
    float acc = bias[c];
#pragma unroll
    for (int dy = 0; dy < 3; ++dy) {
        int yy = y + dy - 1;
        if (yy < 0 || yy > 31) continue;
#pragma unroll
        for (int dx = 0; dx < 3; ++dx) {
            int xx = x + dx - 1;
            if (xx < 0 || xx > 31) continue;
            acc += bf2f(ip[yy * 32 + xx]) * wp[dy * 3 + dx];
        }
    }
    float cl = fminf(fmaxf(acc + 3.0f, 0.0f), 6.0f);
    out[(size_t)bc * HW + pix] = f2bf(acc * cl * (1.0f / 6.0f));
}

// ----------------------- q/k/v prep (relu, transpose, pad) -----------------
__global__ __launch_bounds__(256)
void prepare_qkv_kernel(const unsigned short* __restrict__ qkv,  // [B][768][1024]
                        const unsigned short* __restrict__ agg,  // [B][768][1024]
                        unsigned short* __restrict__ qb,         // [B*16][1024][32]
                        unsigned short* __restrict__ kb,
                        unsigned short* __restrict__ vb) {       // [B*16][1024][48]
    int idx = blockIdx.x * 256 + threadIdx.x;    // 16*16*1024 threads
    int l  = idx & 1023;
    int bn = idx >> 10;
    int b  = bn >> 4, n = bn & 15;
    const unsigned short* src = (n < 8)
        ? qkv + ((size_t)b * 768 + n * 96) * HW + l
        : agg + ((size_t)b * 768 + (n - 8) * 96) * HW + l;
    unsigned short* qo = qb + ((size_t)bn * HW + l) * 32;
    unsigned short* ko = kb + ((size_t)bn * HW + l) * 32;
    unsigned short* vo = vb + ((size_t)bn * HW + l) * 48;
#pragma unroll 4
    for (int d = 0; d < 32; ++d) {
        unsigned short uq = src[(size_t)d * HW];
        unsigned short uk = src[(size_t)(32 + d) * HW];
        qo[d] = (uq & 0x8000u) ? (unsigned short)0 : uq;   // relu in bf16
        ko[d] = (uk & 0x8000u) ? (unsigned short)0 : uk;
        vo[d] = src[(size_t)(64 + d) * HW];
    }
    vo[32] = 0x3F80;                                       // ones column
#pragma unroll
    for (int e = 33; e < 48; ++e) vo[e] = 0;
}

// ------------------- kv = k^T v : M=32, N=48, K=1024 (WMMA) ----------------
__global__ __launch_bounds__(192)
void kv_gemm_kernel(const unsigned short* __restrict__ kb,  // [BN][1024][32]
                    const unsigned short* __restrict__ vb,  // [BN][1024][48]
                    float* __restrict__ kvf,                // [BN][32][48]
                    unsigned short* __restrict__ kvb) {
    __shared__ unsigned short Kst[32][34];   // [d][l]  (A transposed in LDS)
    __shared__ unsigned short Vst[48][34];   // [e][l]  (B cols, K-pairs contiguous)
    const int tid = threadIdx.x, wid = tid >> 5, lane = tid & 31;
    const int hi = lane >> 4, r = lane & 15;
    const int mt = wid / 3, nt = wid % 3;
    const int bn = blockIdx.x;
    const unsigned short* kp = kb + (size_t)bn * HW * 32;
    const unsigned short* vp = vb + (size_t)bn * HW * 48;
    v8f acc = {};
    for (int lb = 0; lb < HW; lb += 32) {
        __syncthreads();
        for (int e = tid; e < 1024; e += 192) {
            int l = e >> 5, d = e & 31;
            Kst[d][l] = kp[(size_t)(lb + l) * 32 + d];
        }
        for (int e = tid; e < 1536; e += 192) {
            int l = e / 48, ee = e % 48;
            Vst[ee][l] = vp[(size_t)(lb + l) * 48 + ee];
        }
        __syncthreads();
        v8u au, bu;
#pragma unroll
        for (int p = 0; p < 8; ++p) {
            int k = kpair(p, hi);
            au[p] = *(const unsigned int*)&Kst[mt * 16 + r][k];
            bu[p] = *(const unsigned int*)&Vst[nt * 16 + r][k];
        }
        acc = __builtin_amdgcn_wmma_f32_16x16x32_bf16(false, __builtin_bit_cast(v16bf, au),
                                                      false, __builtin_bit_cast(v16bf, bu),
                                                      (short)0, acc, false, false);
    }
#pragma unroll
    for (int i = 0; i < 8; ++i) {
        int d = mt * 16 + i + 8 * hi;
        int e = nt * 16 + r;
        size_t o = ((size_t)bn * 32 + d) * 48 + e;
        kvf[o] = acc[i];
        kvb[o] = f2bf(acc[i]);
    }
}

// --------------- out = q*kv, normalize, -> attn layout (WMMA) --------------
__global__ __launch_bounds__(256)
void attn_out_kernel(const unsigned short* __restrict__ qb,   // [BN][1024][32]
                     const unsigned short* __restrict__ kvb,  // [BN][32][48]
                     const float* __restrict__ kvf,
                     unsigned short* __restrict__ attn) {     // [B][512][1024]
    __shared__ unsigned short Qs[128][34];   // [l_local][d]
    __shared__ unsigned short Bt[32][34];    // [e][d]  (K-pairs contiguous)
    __shared__ float kvcol[32];              // kv[:,32] fp32 for denominator
    const int tid = threadIdx.x, wid = tid >> 5, lane = tid & 31;
    const int hi = lane >> 4, r = lane & 15;
    const int bn = blockIdx.y;
    const int b = bn >> 4, n = bn & 15;
    const int l0 = blockIdx.x * 128;
    const unsigned short* qp = qb + ((size_t)bn * HW + l0) * 32;

    {   // kvb tile (e<32 only), 4 elems/thread
        int e0 = tid * 4;
        int d = e0 >> 5, e = e0 & 31;
#pragma unroll
        for (int j = 0; j < 4; ++j)
            Bt[e + j][d] = kvb[((size_t)bn * 32 + d) * 48 + e + j];
    }
    if (tid < 32) kvcol[tid] = kvf[((size_t)bn * 32 + tid) * 48 + 32];
    {   // q tile 128x32, 16 elems/thread
        int e0 = tid * 16;
        int l = e0 >> 5, d0 = e0 & 31;
#pragma unroll
        for (int j = 0; j < 16; ++j)
            Qs[l][d0 + j] = qp[(size_t)l * 32 + d0 + j];
    }
    __syncthreads();

    v8u au, b0u, b1u;
#pragma unroll
    for (int p = 0; p < 8; ++p) {
        int k = kpair(p, hi);
        au[p]  = *(const unsigned int*)&Qs[wid * 16 + r][k];
        b0u[p] = *(const unsigned int*)&Bt[r][k];
        b1u[p] = *(const unsigned int*)&Bt[16 + r][k];
    }
    v16bf a = __builtin_bit_cast(v16bf, au);
    v8f acc0 = {}, acc1 = {};
    acc0 = __builtin_amdgcn_wmma_f32_16x16x32_bf16(false, a, false,
              __builtin_bit_cast(v16bf, b0u), (short)0, acc0, false, false);
    acc1 = __builtin_amdgcn_wmma_f32_16x16x32_bf16(false, a, false,
              __builtin_bit_cast(v16bf, b1u), (short)0, acc1, false, false);

#pragma unroll
    for (int i = 0; i < 8; ++i) {
        int lr = wid * 16 + i + 8 * hi;
        float den = 0.f;
#pragma unroll 8
        for (int d = 0; d < 32; ++d) den += bf2f(Qs[lr][d]) * kvcol[d];
        float inv = 1.0f / (den + 1e-15f);
        int l = l0 + lr;
        attn[((size_t)b * 512 + n * 32 + r) * HW + l]      = f2bf(acc0[i] * inv);
        attn[((size_t)b * 512 + n * 32 + 16 + r) * HW + l] = f2bf(acc1[i] * inv);
    }
}

// ---------------------------------------------------------------------------
extern "C" void kernel_launch(void* const* d_in, const int* in_sizes, int n_in,
                              void* d_out, int out_size, void* d_ws, size_t ws_size,
                              hipStream_t stream) {
    const float* x        = (const float*)d_in[0];
    const float* qkv_w    = (const float*)d_in[1];
    const float* agg_dw_w = (const float*)d_in[2];
    const float* agg_pw_w = (const float*)d_in[3];
    const float* proj_w   = (const float*)d_in[4];
    const float* pg = (const float*)d_in[5], *pb = (const float*)d_in[6];
    const float* pm = (const float*)d_in[7], *pv = (const float*)d_in[8];
    const float* mb1_w = (const float*)d_in[9], *mb1_b = (const float*)d_in[10];
    const float* dw_w  = (const float*)d_in[11], *dw_b = (const float*)d_in[12];
    const float* pw_w  = (const float*)d_in[13];
    const float* wg = (const float*)d_in[14], *wb = (const float*)d_in[15];
    const float* wm = (const float*)d_in[16], *wv = (const float*)d_in[17];

    char* ws = (char*)d_ws;
    // -------- workspace layout (bytes); lifetimes allow the aliases below --
    unsigned short* wqkv = (unsigned short*)(ws + 0);             // 768*256
    unsigned short* wproj = (unsigned short*)(ws + 393216);       // 256*512
    unsigned short* wmb1 = (unsigned short*)(ws + 655360);        // 1024*256
    unsigned short* wpw  = (unsigned short*)(ws + 1179648);       // 256*1024
    unsigned short* wgpw = (unsigned short*)(ws + 1703936);       // 768*32
    unsigned short* xbf   = (unsigned short*)(ws + 1753088);      // 8.39 MB
    unsigned short* qkvbf = (unsigned short*)(ws + 10141696);     // 25.2 MB
    unsigned short* agg2  = (unsigned short*)(ws + 35307520);     // 25.2 MB
    unsigned short* agg1  = (unsigned short*)(ws + 60473344);     // 25.2 MB
    unsigned short* kbuf  = (unsigned short*)(ws + 85639168);     // 16.8 MB
    unsigned short* vbuf  = (unsigned short*)(ws + 102416384);    // 25.2 MB
    float*          kvf   = (float*)(ws + 127582208);             // 1.57 MB
    unsigned short* kvb   = (unsigned short*)(ws + 129155072);    // 0.79 MB
    // aliases (prior tenants dead by the time these are written):
    unsigned short* qbuf  = agg1;                       // q, after agg1 consumed
    unsigned short* attnb = kbuf;                       // attn, after k consumed
    unsigned short* x1bf  = xbf;                        // after x_bf16 consumed
    float*          x1f   = (float*)vbuf;               // after v consumed
    unsigned short* y1    = qkvbf;                      // 33.5 MB spans qkv+agg2
    unsigned short* y2    = agg1;                       // 33.5 MB spans agg1+attn

    float* out = (float*)d_out;

    // -------- weight / input conversion to bf16 --------
    f2bf_kernel<<<(196608 + 255) / 256, 256, 0, stream>>>(qkv_w, wqkv, 196608);
    f2bf_kernel<<<(131072 + 255) / 256, 256, 0, stream>>>(proj_w, wproj, 131072);
    f2bf_kernel<<<(262144 + 255) / 256, 256, 0, stream>>>(mb1_w, wmb1, 262144);
    f2bf_kernel<<<(262144 + 255) / 256, 256, 0, stream>>>(pw_w, wpw, 262144);
    f2bf_kernel<<<(24576 + 255) / 256, 256, 0, stream>>>(agg_pw_w, wgpw, 24576);
    f2bf_kernel<<<(4194304 + 255) / 256, 256, 0, stream>>>(x, xbf, 4194304);

    // -------- S1: qkv = W_qkv * x  (768x256 GEMM) --------
    gemm_bf16_kernel<0><<<dim3(8, 24, BB), 256, 0, stream>>>(
        wqkv, xbf, 768, 256, qkvbf, nullptr, nullptr, nullptr, nullptr, nullptr, nullptr);
    // -------- S2: depthwise 5x5 --------
    dw5x5_kernel<<<dim3(BB * 768, 4), 256, 0, stream>>>(qkvbf, agg_dw_w, agg1);
    // -------- S3: grouped pointwise (24 groups of 32x32) --------
    gemm_group32_kernel<<<dim3(8, 24, BB), 256, 0, stream>>>(wgpw, agg1, agg2);
    // -------- S4: build q/k/v (relu, transpose, pad ones col) --------
    prepare_qkv_kernel<<<1024, 256, 0, stream>>>(qkvbf, agg2, qbuf, kbuf, vbuf);
    // -------- S5: kv = k^T v  (per b,head; K=1024) --------
    kv_gemm_kernel<<<256, 192, 0, stream>>>(kbuf, vbuf, kvf, kvb);
    // -------- S6: out = q*kv, normalize -> attn (b,512,1024) --------
    attn_out_kernel<<<dim3(8, 256), 256, 0, stream>>>(qbuf, kvb, kvf, attnb);
    // -------- S7: x1 = x + BN(proj * attn)  (256x512 GEMM) --------
    gemm_bf16_kernel<1><<<dim3(8, 8, BB), 256, 0, stream>>>(
        wproj, attnb, 256, 512, x1bf, x1f, x, pg, pb, pm, pv);
    // -------- S8: y1 = hswish(mb1 * x1 + b)  (1024x256 GEMM) --------
    gemm_bf16_kernel<2><<<dim3(8, 32, BB), 256, 0, stream>>>(
        wmb1, x1bf, 1024, 256, y1, nullptr, nullptr, mb1_b, nullptr, nullptr, nullptr);
    // -------- S9: y2 = hswish(dw3x3(y1) + b) --------
    dw3x3_kernel<<<dim3(BB * 1024, 4), 256, 0, stream>>>(y1, dw_w, dw_b, y2);
    // -------- S10: out = x1 + BN(pw * y2)  (256x1024 GEMM) --------
    gemm_bf16_kernel<3><<<dim3(8, 8, BB), 256, 0, stream>>>(
        wpw, y2, 256, 1024, nullptr, out, x1f, wg, wb, wm, wv);

    (void)in_sizes; (void)n_in; (void)out_size; (void)ws_size;
}
